// MambaBlock_39917426049299
// MI455X (gfx1250) — matrix operand order repaired and checked by reading, not compiled, not run on hardware
//
#include <hip/hip_runtime.h>
#include <hip/hip_bf16.h>

// ---------------------------------------------------------------------------
// Problem constants (from reference setup_inputs)
// ---------------------------------------------------------------------------
#define BB    4
#define LL    2048
#define DM    768
#define DIN   1536
#define NST   16
#define KCONV 4
#define MROWS (BB * LL)          // 8192
#define NDBC  (1 + 2 * NST)      // 33

// WMMA tile config: 256 threads = 8 waves in 2(M) x 4(N); each wave owns a
// 16x32 output slab (2 WMMAs per K-step sharing one A fragment).
#define BM 32
#define BN 128
#define BK 32
#define LDSK (BK + 4)            // 36 ushorts -> 72B row stride, 8B aligned

typedef __attribute__((ext_vector_type(16))) __bf16 v16bf;
typedef __attribute__((ext_vector_type(8)))  float  v8f;

// ---------------------------------------------------------------------------
// helpers
// ---------------------------------------------------------------------------
__device__ __forceinline__ unsigned short f32_to_bf16_rne(float f) {
    union { float f; unsigned u; } v; v.f = f;
    unsigned u = v.u;
    u = u + 0x7FFFu + ((u >> 16) & 1u);   // round-to-nearest-even
    return (unsigned short)(u >> 16);
}

__device__ __forceinline__ float block_reduce_sum_256(float val, float* red) {
    int t = threadIdx.x;
    red[t] = val; __syncthreads();
    #pragma unroll
    for (int o = 128; o > 0; o >>= 1) {
        if (t < o) red[t] += red[t + o];
        __syncthreads();
    }
    float r = red[0];
    __syncthreads();
    return r;
}

__device__ __forceinline__ float silu_f(float x) {
    return x / (1.f + __expf(-x));
}
__device__ __forceinline__ float softplus_f(float x) {
    return (x > 20.f) ? x : __logf(1.f + __expf(x));
}

// CDNA5 async global->LDS copy (8 bytes per lane), tracked by ASYNCcnt.
// lds_off: LDS byte offset (low 32 bits of flat shared-space address).
__device__ __forceinline__ void async_copy_b64(unsigned lds_off, const void* gptr) {
    asm volatile("global_load_async_to_lds_b64 %0, %1, off"
                 :: "v"(lds_off), "v"(gptr)
                 : "memory");
}
__device__ __forceinline__ void wait_asynccnt0() {
    asm volatile("s_wait_asynccnt 0" ::: "memory");
}

// ---------------------------------------------------------------------------
// 1) |W| partial sums  ->  2) scale = max(mean,1e-5)  ->  3) ternary bf16
// ---------------------------------------------------------------------------
__global__ void absmean_partial_kernel(const float* __restrict__ W, long n,
                                       float* __restrict__ partial) {
    __shared__ float red[256];
    float s = 0.f;
    for (size_t i = (size_t)blockIdx.x * 256 + threadIdx.x; i < (size_t)n;
         i += (size_t)gridDim.x * 256)
        s += fabsf(W[i]);
    s = block_reduce_sum_256(s, red);
    if (threadIdx.x == 0) partial[blockIdx.x] = s;
}

__global__ void finalize_scale_kernel(const float* __restrict__ partial, int nb,
                                      float inv_n, float* __restrict__ scale) {
    __shared__ float red[256];
    float s = 0.f;
    for (int i = threadIdx.x; i < nb; i += 256) s += partial[i];
    s = block_reduce_sum_256(s, red);
    if (threadIdx.x == 0) *scale = fmaxf(s * inv_n, 1e-5f);
}

__global__ void quantize_w_kernel(const float* __restrict__ W,
                                  const float* __restrict__ scale,
                                  unsigned short* __restrict__ Wq, long n) {
    size_t i = (size_t)blockIdx.x * 256 + threadIdx.x;
    if (i < (size_t)n) {
        float s  = *scale;
        float wn = W[i] / s;
        wn = fminf(fmaxf(wn, -1.f), 1.f);
        Wq[i] = f32_to_bf16_rne(rintf(wn));   // exactly {-1,0,1} in bf16
    }
}

// ---------------------------------------------------------------------------
// RMSNorm (optionally double-normed: rmsnorm(rmsnorm(x,w1),w2)) -> bf16
// One block (256 threads) per row.
// ---------------------------------------------------------------------------
__global__ void rmsnorm_bf16_kernel(const float* __restrict__ x,
                                    const float* __restrict__ w1,
                                    const float* __restrict__ w2,   // null => single
                                    unsigned short* __restrict__ out,
                                    int D, int in_stride) {
    __shared__ float red[256];
    size_t row = blockIdx.x;
    const float* xr = x + row * (size_t)in_stride;

    float v[8];
    int cnt = 0;
    float ss = 0.f;
    for (int i = threadIdx.x; i < D; i += 256) {
        float t = xr[i];
        v[cnt++] = t;
        ss += t * t;
    }
    float ms = block_reduce_sum_256(ss, red) / (float)D;
    float rs = rsqrtf(ms + 1e-6f);

    if (w2 != nullptr) {
        float ss2 = 0.f;
        int c = 0;
        for (int i = threadIdx.x; i < D; i += 256, ++c) {
            float h = v[c] * rs * w1[i];
            v[c] = h;
            ss2 += h * h;
        }
        float ms2 = block_reduce_sum_256(ss2, red) / (float)D;
        float rs2 = rsqrtf(ms2 + 1e-6f);
        c = 0;
        for (int i = threadIdx.x; i < D; i += 256, ++c)
            out[row * (size_t)D + i] = f32_to_bf16_rne(v[c] * rs2 * w2[i]);
    } else {
        int c = 0;
        for (int i = threadIdx.x; i < D; i += 256, ++c)
            out[row * (size_t)D + i] = f32_to_bf16_rne(v[c] * rs * w1[i]);
    }
    (void)cnt;
}

// ---------------------------------------------------------------------------
// bf16 WMMA GEMM:  C[M][N] = (A[M][K] @ Bw[N][K]^T) * (*scale) (+ residual)
// A, Bw: bf16 bits (ushort), row-major. M%32==0, N%128==0, K%32==0.
// Double-buffered LDS; tiles staged with GLOBAL_LOAD_ASYNC_TO_LDS_B64 so the
// next K-step's copies overlap the current step's WMMAs. Fragment layouts per
// CDNA5 ISA 7.12.2 (wave32).
// ---------------------------------------------------------------------------
__global__ void gemm_bf16_wmma_kernel(const unsigned short* __restrict__ A,
                                      const unsigned short* __restrict__ Bw,
                                      float* __restrict__ C,
                                      const float* __restrict__ scale_ptr,
                                      const float* __restrict__ residual,
                                      int M, int N, int K) {
    __shared__ unsigned short As[2][BM][LDSK];
    __shared__ unsigned short Bs[2][BN][LDSK];

    const int tid  = threadIdx.x;
    const int wave = tid >> 5;
    const int lane = tid & 31;
    const int wm = wave >> 2;        // 0..1  (16-row slab)
    const int wn = wave & 3;         // 0..3  (32-col slab)
    const int bm = blockIdx.y * BM;
    const int bn = blockIdx.x * BN;

    const int half = lane >> 4;      // 0: lanes 0-15, 1: lanes 16-31
    const int lr   = lane & 15;

    // staging coords: each lane moves 4 consecutive bf16 (8B) per chunk
    const int ar  = tid >> 3;            // 0..31 row in A tile
    const int ac4 = (tid & 7) << 2;      // col group 0,4,...,28

    // LDS byte offsets (low 32 bits of flat shared-space address)
    const unsigned asb0 = (unsigned)(size_t)&As[0][0][0];
    const unsigned asb1 = (unsigned)(size_t)&As[1][0][0];
    const unsigned bsb0 = (unsigned)(size_t)&Bs[0][0][0];
    const unsigned bsb1 = (unsigned)(size_t)&Bs[1][0][0];

    v8f acc0 = {0.f, 0.f, 0.f, 0.f, 0.f, 0.f, 0.f, 0.f};
    v8f acc1 = {0.f, 0.f, 0.f, 0.f, 0.f, 0.f, 0.f, 0.f};

    // issue one buffer's worth of async copies for K-offset k0
    auto stage = [&](int k0, unsigned asb, unsigned bsb) {
        async_copy_b64(asb + (unsigned)(ar * LDSK + ac4) * 2,
                       &A[(size_t)(bm + ar) * K + (k0 + ac4)]);
        #pragma unroll
        for (int i = 0; i < 4; ++i) {
            int e  = tid + i * 256;
            int r  = e >> 3;
            int c4 = (e & 7) << 2;
            async_copy_b64(bsb + (unsigned)(r * LDSK + c4) * 2,
                           &Bw[(size_t)(bn + r) * K + (k0 + c4)]);
        }
    };

    const int nIter = K / BK;
    stage(0, asb0, bsb0);                       // prologue: fill buffer 0

    for (int it = 0; it < nIter; ++it) {
        const int cur = it & 1;
        wait_asynccnt0();                       // my copies into buf[cur] done
        __syncthreads();                        // everyone's copies visible

        if (it + 1 < nIter)                     // prefetch next tile into other
            stage((it + 1) * BK, cur ? asb0 : asb1, cur ? bsb0 : bsb1);

        const unsigned short (*Ac)[LDSK] = As[cur];
        const unsigned short (*Bc)[LDSK] = Bs[cur];

        // A fragment: 16x32, lane holds row (lane&15); VGPR v -> K pair
        union { v16bf v; unsigned u[8]; } fa, fb0, fb1;
        const int am = wm * 16 + lr;
        #pragma unroll
        for (int vv = 0; vv < 8; ++vv) {
            int kb = ((vv >> 2) << 4) + (half << 3) + ((vv & 3) << 1);
            fa.u[vv] = *(const unsigned*)&Ac[am][kb];
        }
        // B fragments: cols n = wn*32 + lr and +16; VGPR v -> K = (v>>2)*16+half*8+2(v&3)
        const int b0 = wn * 32 + lr;
        #pragma unroll
        for (int vv = 0; vv < 8; ++vv) {
            int kb = ((vv >> 2) << 4) + (half << 3) + ((vv & 3) << 1);
            fb0.u[vv] = *(const unsigned*)&Bc[b0][kb];
            fb1.u[vv] = *(const unsigned*)&Bc[b0 + 16][kb];
        }

        acc0 = __builtin_amdgcn_wmma_f32_16x16x32_bf16(
                 false, fa.v, false, fb0.v, (short)0, acc0, false, false);
        acc1 = __builtin_amdgcn_wmma_f32_16x16x32_bf16(
                 false, fa.v, false, fb1.v, (short)0, acc1, false, false);
    }

    // epilogue: C/D 16x16 f32 layout — VGPR v: row = v + half*8, col = lane&15
    const float s = *scale_ptr;
    const int gn0 = bn + wn * 32 + lr;
    #pragma unroll
    for (int vv = 0; vv < 8; ++vv) {
        int gm = bm + wm * 16 + half * 8 + vv;
        size_t rowoff = (size_t)gm * N;
        float v0 = acc0[vv] * s;
        float v1 = acc1[vv] * s;
        if (residual) {
            v0 += residual[rowoff + gn0];
            v1 += residual[rowoff + gn0 + 16];
        }
        C[rowoff + gn0]      = v0;
        C[rowoff + gn0 + 16] = v1;
    }
}

// ---------------------------------------------------------------------------
// Depthwise causal conv (K=4) + bias + SiLU.  x_path lives in xz[..., :DIN].
// ---------------------------------------------------------------------------
__global__ void conv_silu_kernel(const float* __restrict__ xz,
                                 const float* __restrict__ conv_w,
                                 const float* __restrict__ conv_b,
                                 float* __restrict__ xc) {
    size_t idx = (size_t)blockIdx.x * 256 + threadIdx.x;
    if (idx >= (size_t)MROWS * DIN) return;
    int d = (int)(idx % DIN);
    int l = (int)((idx / DIN) % LL);
    int b = (int)(idx / ((size_t)DIN * LL));

    float s = conv_b[d];
    #pragma unroll
    for (int k = 0; k < KCONV; ++k) {
        int ls = l - (KCONV - 1) + k;
        if (ls >= 0)
            s = fmaf(xz[((size_t)(b * LL + ls)) * (2 * DIN) + d],
                     conv_w[d * KCONV + k], s);
    }
    xc[idx] = silu_f(s);
}

// ---------------------------------------------------------------------------
// dbc[m][0:33] = xc[m][:] @ W_x.T   (tiny N, one block per row)
// ---------------------------------------------------------------------------
__global__ void dbc_kernel(const float* __restrict__ xc,
                           const float* __restrict__ Wx,
                           float* __restrict__ dbc) {
    __shared__ float xs[DIN];
    size_t m = blockIdx.x;
    const float* xr = xc + m * (size_t)DIN;
    for (int i = threadIdx.x; i < DIN; i += 64) xs[i] = xr[i];
    __syncthreads();
    int j = threadIdx.x;
    if (j < NDBC) {
        const float* wr = Wx + (size_t)j * DIN;
        float acc = 0.f;
        for (int k = 0; k < DIN; ++k) acc = fmaf(xs[k], wr[k], acc);
        dbc[m * NDBC + j] = acc;
    }
}

// ---------------------------------------------------------------------------
// Selective scan. One thread per (b,d) channel; N=16 state in registers.
// Writes y = (scan + u*D) * silu(z) over the x-path half of xz.
// grid = BB * (DIN/256) = 24 blocks.
// ---------------------------------------------------------------------------
__global__ void scan_kernel(const float* __restrict__ dbc,
                            const float* __restrict__ xc,
                            float* __restrict__ xz,
                            const float* __restrict__ A_log,
                            const float* __restrict__ dt_w,
                            const float* __restrict__ dt_b,
                            const float* __restrict__ D_param) {
    __shared__ float s[NDBC];
    const int bpb = DIN / 256;                       // blocks per batch = 6
    const int b = blockIdx.x / bpb;
    const int d = (blockIdx.x % bpb) * 256 + threadIdx.x;

    float A[NST];
    #pragma unroll
    for (int n = 0; n < NST; ++n) A[n] = -__expf(A_log[d * NST + n]);
    float h[NST];
    #pragma unroll
    for (int n = 0; n < NST; ++n) h[n] = 0.f;

    const float dtw = dt_w[d], dtb = dt_b[d], Dp = D_param[d];

    for (int l = 0; l < LL; ++l) {
        if (threadIdx.x < NDBC)
            s[threadIdx.x] = dbc[((size_t)(b * LL + l)) * NDBC + threadIdx.x];
        __syncthreads();

        float delta = softplus_f(s[0] * dtw + dtb);
        size_t rowc = (size_t)(b * LL + l) * DIN + d;
        float u  = xc[rowc];
        float du = delta * u;

        float y = 0.f;
        #pragma unroll
        for (int n = 0; n < NST; ++n) {
            float dA = __expf(delta * A[n]);
            h[n] = fmaf(dA, h[n], du * s[1 + n]);
            y    = fmaf(h[n], s[1 + NST + n], y);
        }

        size_t rowz = (size_t)(b * LL + l) * (2 * DIN);
        float z = xz[rowz + DIN + d];
        xz[rowz + d] = (y + u * Dp) * silu_f(z);
        __syncthreads();
    }
}

// ---------------------------------------------------------------------------
// host-side launcher
// ---------------------------------------------------------------------------
extern "C" void kernel_launch(void* const* d_in, const int* in_sizes, int n_in,
                              void* d_out, int out_size, void* d_ws, size_t ws_size,
                              hipStream_t stream) {
    const float* x         = (const float*)d_in[0];
    const float* norm_w    = (const float*)d_in[1];
    const float* in_norm_w = (const float*)d_in[2];
    const float* W_in      = (const float*)d_in[3];
    const float* conv_w    = (const float*)d_in[4];
    const float* conv_b    = (const float*)d_in[5];
    const float* W_x       = (const float*)d_in[6];
    const float* dt_w      = (const float*)d_in[7];
    const float* dt_b      = (const float*)d_in[8];
    const float* A_log     = (const float*)d_in[9];
    const float* D_param   = (const float*)d_in[10];
    const float* out_nw    = (const float*)d_in[11];
    const float* W_out     = (const float*)d_in[12];
    float* out = (float*)d_out;

    const long  nWin  = (long)(2 * DIN) * DM;   // 3072*768
    const long  nWout = (long)DM * DIN;         // 768*1536

    // workspace carve-out (256B aligned)
    size_t off = 0;
    auto carve = [&](size_t bytes) {
        size_t o = off;
        off = (off + bytes + 255) & ~(size_t)255;
        return o;
    };
    char* ws = (char*)d_ws;
    unsigned short* Wq_in  = (unsigned short*)(ws + carve((size_t)nWin  * 2));
    unsigned short* Wq_out = (unsigned short*)(ws + carve((size_t)nWout * 2));
    unsigned short* Xn     = (unsigned short*)(ws + carve((size_t)MROWS * DM  * 2));
    float*          xz     = (float*)(ws + carve((size_t)MROWS * 2 * DIN * 4));
    float*          xc     = (float*)(ws + carve((size_t)MROWS * DIN * 4));
    float*          dbc    = (float*)(ws + carve((size_t)MROWS * NDBC * 4));
    unsigned short* Yn     = (unsigned short*)(ws + carve((size_t)MROWS * DIN * 2));
    float*          p_in   = (float*)(ws + carve(256 * 4));
    float*          p_out  = (float*)(ws + carve(256 * 4));
    float*          sc_in  = (float*)(ws + carve(4));
    float*          sc_out = (float*)(ws + carve(4));

    // --- weight scales + ternary bf16 quantization -------------------------
    absmean_partial_kernel<<<256, 256, 0, stream>>>(W_in,  nWin,  p_in);
    finalize_scale_kernel<<<1, 256, 0, stream>>>(p_in, 256, 1.f / (float)nWin,  sc_in);
    absmean_partial_kernel<<<256, 256, 0, stream>>>(W_out, nWout, p_out);
    finalize_scale_kernel<<<1, 256, 0, stream>>>(p_out, 256, 1.f / (float)nWout, sc_out);
    quantize_w_kernel<<<(unsigned)((nWin  + 255) / 256), 256, 0, stream>>>(W_in,  sc_in,  Wq_in,  nWin);
    quantize_w_kernel<<<(unsigned)((nWout + 255) / 256), 256, 0, stream>>>(W_out, sc_out, Wq_out, nWout);

    // --- double RMSNorm(x) -> bf16 activations -----------------------------
    rmsnorm_bf16_kernel<<<MROWS, 256, 0, stream>>>(x, norm_w, in_norm_w, Xn, DM, DM);

    // --- xz = Xn @ Wq_in^T * scale_in   (8192 x 3072, K=768) ---------------
    {
        dim3 grid((2 * DIN) / BN, MROWS / BM);
        gemm_bf16_wmma_kernel<<<grid, 256, 0, stream>>>(Xn, Wq_in, xz, sc_in,
                                                        nullptr, MROWS, 2 * DIN, DM);
    }

    // --- causal conv + SiLU ------------------------------------------------
    conv_silu_kernel<<<(unsigned)(((size_t)MROWS * DIN + 255) / 256), 256, 0, stream>>>(
        xz, conv_w, conv_b, xc);

    // --- dbc = xc @ W_x^T (N=33) ------------------------------------------
    dbc_kernel<<<MROWS, 64, 0, stream>>>(xc, W_x, dbc);

    // --- selective scan, fused softplus/D-skip/silu(z); y -> xz[..., :DIN] -
    scan_kernel<<<BB * (DIN / 256), 256, 0, stream>>>(dbc, xc, xz, A_log,
                                                      dt_w, dt_b, D_param);

    // --- RMSNorm(y) -> bf16 (y rows live at stride 2*DIN inside xz) --------
    rmsnorm_bf16_kernel<<<MROWS, 256, 0, stream>>>(xz, out_nw, nullptr, Yn,
                                                   DIN, 2 * DIN);

    // --- out = Yn @ Wq_out^T * scale_out + x  (8192 x 768, K=1536) ---------
    {
        dim3 grid(DM / BN, MROWS / BM);
        gemm_bf16_wmma_kernel<<<grid, 256, 0, stream>>>(Yn, Wq_out, out, sc_out,
                                                        x, MROWS, DM, DIN);
    }
    (void)in_sizes; (void)n_in; (void)out_size; (void)ws_size;
}
